// HyperGraphAutoEncoder_21655225107258
// MI455X (gfx1250) — matrix operand order repaired and checked by reading, not compiled, not run on hardware
//
#include <hip/hip_runtime.h>
#include <hip/hip_bf16.h>

typedef __attribute__((ext_vector_type(16))) _Float16 v16h;
typedef __attribute__((ext_vector_type(8)))  float    v8f;
typedef __attribute__((ext_vector_type(4)))  unsigned int u32x4;
typedef __attribute__((ext_vector_type(8)))  int      i32x8;
typedef __attribute__((ext_vector_type(4)))  int      i32x4;

#define NODES 6144

#ifndef __has_builtin
#define __has_builtin(x) 0
#endif
#if __has_builtin(__builtin_amdgcn_tensor_load_to_lds)
#define HAS_TDM 1
#else
#define HAS_TDM 0
#endif

#if HAS_TDM
// ---------------------------------------------------------------------------
// TDM: DMA a 16-row x 32-col f32 tile of a NODESxNODES row-major matrix into
// LDS (contiguous 16x32 layout).  D# per CDNA5 ISA ch.8:
//   group0: count=1 | lds_addr | global_addr(57b) | type=2
//   group1: data_size=4B, tensor_dim0/1 = NODES, tile = 32x16,
//           tensor_dim0_stride = NODES
//   groups 2/3: zero (2D tensor)
// ---------------------------------------------------------------------------
__device__ __forceinline__ void tdm_load_tile(const float* __restrict__ gsrc,
                                              unsigned lds_byte_off) {
    unsigned long long ga = (unsigned long long)(uintptr_t)gsrc;
    u32x4 g0;
    g0[0] = 1u;                                            // count=1, user desc
    g0[1] = lds_byte_off;                                  // LDS dest (bytes)
    g0[2] = (unsigned)(ga & 0xFFFFFFFFu);                  // global addr lo
    g0[3] = (unsigned)((ga >> 32) & 0x01FFFFFFu) | (2u << 30); // addr hi | type=2

    const unsigned ND = (unsigned)NODES;                   // 6144 (< 2^16)
    i32x8 g1;
    g1[0] = (int)(2u << 16);                  // wg_mask=0, data_size=2 (4B)
    g1[1] = (int)((ND & 0xFFFFu) << 16);      // tensor_dim0[15:0]  -> bits 63:48
    g1[2] = (int)(((ND >> 16) & 0xFFFFu) | ((ND & 0xFFFFu) << 16)); // d0 hi | d1 lo
    g1[3] = (int)(((ND >> 16) & 0xFFFFu) | (32u << 16));   // d1 hi | tile_dim0=32
    g1[4] = (int)16u;                          // tile_dim1=16, tile_dim2=0
    g1[5] = (int)ND;                           // tensor_dim0_stride lo32
    g1[6] = 0;                                 // stride hi16 | dim1_stride lo16
    g1[7] = 0;
    i32x4 gz = {0, 0, 0, 0};
#if __clang_major__ >= 23
    i32x8 gz8 = {0, 0, 0, 0, 0, 0, 0, 0};
    __builtin_amdgcn_tensor_load_to_lds(g0, g1, gz, gz, gz8, 0);
#else
    __builtin_amdgcn_tensor_load_to_lds(g0, g1, gz, gz, 0);
#endif
}
#endif // HAS_TDM

// ---------------------------------------------------------------------------
// 1) expmap0: x (N,128) -> h0 (N,129) = [cosh(||x||), sinh(||x||)*x/||x||]
// ---------------------------------------------------------------------------
__global__ void expmap_kernel(const float* __restrict__ x, float* __restrict__ h0) {
    int row = blockIdx.x;
    int c = threadIdx.x;              // 128 threads
    __shared__ float red[128];
    float v = x[(size_t)row * 128 + c];
    red[c] = v * v;
    __syncthreads();
    for (int s = 64; s > 0; s >>= 1) {
        if (c < s) red[c] += red[c + s];
        __syncthreads();
    }
    float nrm = fmaxf(sqrtf(red[0]), 1e-8f);
    if (c == 0) h0[(size_t)row * 129] = coshf(nrm);
    h0[(size_t)row * 129 + 1 + c] = sinhf(nrm) * v / nrm;
}

// ---------------------------------------------------------------------------
// 2) lorentz_linear: y = (relu?)in @ W + b ; time = sigmoid(y0)*s + 1.1 ;
//    space /= sqrt(max(sum(space^2)/max(time^2-1,eps),eps))
//    Writes result transposed in f16: gT[col*N + row]  (cols [OUT,OUTP) = 0)
// ---------------------------------------------------------------------------
__global__ void lin_norm_kernel(const float* __restrict__ in, int IN, int OUT, int OUTP,
                                const float* __restrict__ W, const float* __restrict__ b,
                                const float* __restrict__ s_ptr, int do_relu,
                                _Float16* __restrict__ gT) {
    int row = blockIdx.x;
    int c = threadIdx.x;              // 128 threads
    __shared__ float xin[129];
    __shared__ float red[128];
    __shared__ float sh_time, sh_denom;
    for (int k = c; k < IN; k += 128) {
        float v = in[(size_t)row * IN + k];
        if (do_relu) v = fmaxf(v, 0.f);
        xin[k] = v;
    }
    __syncthreads();
    float acc = 0.f;
    if (c < OUT) {
        acc = b[c];
        for (int k = 0; k < IN; ++k) acc = fmaf(xin[k], W[(size_t)k * OUT + c], acc);
    }
    red[c] = (c >= 1 && c < OUT) ? acc * acc : 0.f;
    __syncthreads();
    for (int s = 64; s > 0; s >>= 1) {
        if (c < s) red[c] += red[c + s];
        __syncthreads();
    }
    if (c == 0) {
        float sval = *s_ptr;
        float time = 1.f / (1.f + __expf(-acc)) * sval + 1.1f;
        float denom = sqrtf(fmaxf(red[0] / fmaxf(time * time - 1.f, 1e-8f), 1e-8f));
        sh_time = time;
        sh_denom = denom;
    }
    __syncthreads();
    float outv = 0.f;
    if (c == 0)           outv = sh_time;
    else if (c < OUT)     outv = acc / sh_denom;
    if (c < OUTP) gT[(size_t)c * NODES + row] = (_Float16)outv;
}

// ---------------------------------------------------------------------------
// 3) aggregation: m = lorentz_agg(adj @ g).  One block = 16 output rows,
//    NT waves (one 16x16 WMMA column tile each).  The shared 16x32 f32 adj
//    tile is streamed into LDS by the Tensor Data Mover, double-buffered
//    against the WMMA k-loop (TENSORcnt).  Row normalization fused via LDS.
// ---------------------------------------------------------------------------
__global__ void agg_kernel(const float* __restrict__ adj, const _Float16* __restrict__ gT,
                           float* __restrict__ m, int OUT) {
    const int N = NODES;
    __shared__ __align__(16) float abuf[2][16 * 32];
    __shared__ float tile[16][128];
    __shared__ float denoms[16];

    int rowbase = blockIdx.x * 16;
    int wave = threadIdx.x >> 5;
    int lane = threadIdx.x & 31;
    int lm = lane & 15;
    int hiA = (lane >> 4) * 8;    // A-fragment K split per lane half
    int hiB = (lane >> 4) * 16;   // unused; B fragment indexes with hiA too

    const float*    abase = adj + (size_t)rowbase * N;
    const _Float16* bcol  = gT  + (size_t)(wave * 16 + lm) * N;

    const int nIt = N / 32;
    v8f c = {};
#if HAS_TDM
    if (threadIdx.x < 32)
        tdm_load_tile(abase, (unsigned)(size_t)&abuf[0][0]);
#endif
    for (int it = 0; it < nIt; ++it) {
        int cur = it & 1;
        int kb = it * 32;
#if HAS_TDM
        if (threadIdx.x < 32) {
            if (it + 1 < nIt) {
                tdm_load_tile(abase + (kb + 32),
                              (unsigned)(size_t)&abuf[cur ^ 1][0]);
                __builtin_amdgcn_s_wait_tensorcnt(1);   // oldest tile landed
            } else {
                __builtin_amdgcn_s_wait_tensorcnt(0);
            }
        }
#else
        for (int idx = threadIdx.x; idx < 16 * 32; idx += blockDim.x)
            abuf[cur][idx] = abase[(size_t)(idx >> 5) * N + kb + (idx & 31)];
#endif
        __syncthreads();              // tile[cur] visible to all waves
        v16h a, bf;
#pragma unroll
        for (int i = 0; i < 8; ++i) {
            a[i]     = (_Float16)abuf[cur][lm * 32 + hiA + i];
            a[8 + i] = (_Float16)abuf[cur][lm * 32 + 16 + hiA + i];
        }
#pragma unroll
        for (int i = 0; i < 16; ++i) bf[i] = bcol[kb + hiA + (i & 7) + ((i >> 3) << 4)];
        c = __builtin_amdgcn_wmma_f32_16x16x32_f16(false, a, false, bf,
                                                   (short)0, c, false, false);
        __syncthreads();              // reads done before buffer reuse
    }

    int roff = (lane >= 16) ? 8 : 0;
#pragma unroll
    for (int r = 0; r < 8; ++r) tile[r + roff][wave * 16 + lm] = c[r];
    __syncthreads();

    if (threadIdx.x < 16) {
        int rr = threadIdx.x;
        float inner = 0.f;
        for (int cc = 1; cc < OUT; ++cc) inner += tile[rr][cc] * tile[rr][cc];
        float t0 = tile[rr][0];
        inner -= t0 * t0;
        denoms[rr] = sqrtf(fmaxf(-inner, 1e-8f));
    }
    __syncthreads();

    int total = 16 * OUT;
    for (int idx = threadIdx.x; idx < total; idx += blockDim.x) {
        int rr = idx / OUT, cc = idx % OUT;
        m[(size_t)(rowbase + rr) * OUT + cc] = tile[rr][cc] / denoms[rr];
    }
}

// ---------------------------------------------------------------------------
// 4) Poincare clip + lift back: m2 (N,65) -> z (N,65) f32 (output tail),
//    zs (N,64) f16, z0 (N) f32
// ---------------------------------------------------------------------------
__global__ void poincare_kernel(const float* __restrict__ m2, const float* __restrict__ scale_p,
                                float* __restrict__ z_out, _Float16* __restrict__ zs,
                                float* __restrict__ z0) {
    int row = blockIdx.x;
    int c = threadIdx.x;              // 64 threads
    __shared__ float red[64];
    float time = m2[(size_t)row * 65];
    float p = m2[(size_t)row * 65 + 1 + c] / (time + 1.f);
    red[c] = p * p;
    __syncthreads();
    for (int s = 32; s > 0; s >>= 1) { if (c < s) red[c] += red[c + s]; __syncthreads(); }
    float nrm = fmaxf(sqrtf(red[0]), 1e-8f);
    float sc = fminf(fmaxf(*scale_p, 0.01f), 0.999f);
    p = p / nrm * sc;
    __syncthreads();
    red[c] = p * p;
    __syncthreads();
    for (int s = 32; s > 0; s >>= 1) { if (c < s) red[c] += red[c + s]; __syncthreads(); }
    float n2 = red[0];
    float den = fmaxf(1.f - n2, 1e-8f);
    float zsv = 2.f * p / den;
    if (c == 0) {
        float zt = (1.f + n2) / den;
        z_out[(size_t)row * 65] = zt;
        z0[row] = zt;
    }
    z_out[(size_t)row * 65 + 1 + c] = zsv;
    zs[(size_t)row * 64 + c] = (_Float16)zsv;
}

// ---------------------------------------------------------------------------
// 5) Gram decode: G = zs zs^T - z0 z0^T ; adj_pred = sigmoid(sigmoid((2-d^2)))
//    8 waves/block -> 32x64 output tile; K=64 = two WMMA steps.
// ---------------------------------------------------------------------------
__global__ void gram_kernel(const _Float16* __restrict__ zs, const float* __restrict__ z0,
                            float* __restrict__ out) {
    const int N = NODES;
    int wave = threadIdx.x >> 5;
    int lane = threadIdx.x & 31;
    int trow = wave >> 2, tcol = wave & 3;
    int rowbase = blockIdx.y * 32 + trow * 16;
    int colbase = blockIdx.x * 64 + tcol * 16;
    int lm = lane & 15;
    int hiA = (lane >> 4) * 8;
    int hiB = (lane >> 4) * 16;

    const _Float16* arow = zs + (size_t)(rowbase + lm) * 64;
    const _Float16* brow = zs + (size_t)(colbase + lm) * 64;

    v8f c = {};
#pragma unroll
    for (int kb = 0; kb < 64; kb += 32) {
        v16h a, b;
#pragma unroll
        for (int i = 0; i < 8; ++i) {
            a[i]     = arow[kb + hiA + i];
            a[8 + i] = arow[kb + 16 + hiA + i];
        }
#pragma unroll
        for (int i = 0; i < 16; ++i) b[i] = brow[kb + hiA + (i & 7) + ((i >> 3) << 4)];
        c = __builtin_amdgcn_wmma_f32_16x16x32_f16(false, a, false, b,
                                                   (short)0, c, false, false);
    }

    int col = colbase + lm;
    float z0c = z0[col];
    int roff = (lane >= 16) ? 8 : 0;
#pragma unroll
    for (int r = 0; r < 8; ++r) {
        int row = rowbase + r + roff;
        float g = c[r] - z0[row] * z0c;
        float xx = fmaxf(-g, 1.f + 1e-7f);
        float d  = __logf(xx + sqrtf(xx * xx - 1.f));
        float pr = 1.f / (1.f + __expf(-(2.f - d * d)));   // (R - d^2)/T, R=2 T=1
        float ap = 1.f / (1.f + __expf(-pr));
        out[(size_t)row * N + col] = ap;
    }
}

// ---------------------------------------------------------------------------
extern "C" void kernel_launch(void* const* d_in, const int* in_sizes, int n_in,
                              void* d_out, int out_size, void* d_ws, size_t ws_size,
                              hipStream_t stream) {
    const int N = NODES;
    const float* x     = (const float*)d_in[0];
    const float* adj   = (const float*)d_in[1];
    const float* W0    = (const float*)d_in[2];
    const float* b0    = (const float*)d_in[3];
    const float* s0    = (const float*)d_in[4];
    const float* W1    = (const float*)d_in[5];
    const float* b1    = (const float*)d_in[6];
    const float* s1    = (const float*)d_in[7];
    const float* W2    = (const float*)d_in[8];
    const float* b2    = (const float*)d_in[9];
    const float* s2    = (const float*)d_in[10];
    const float* scale = (const float*)d_in[11];
    float* out = (float*)d_out;

    char* ws = (char*)d_ws;
    size_t off = 0;
    float*    h0 = (float*)   (ws + off); off += (size_t)N * 129 * 4;   // 3,170,304
    _Float16* gT = (_Float16*)(ws + off); off += (size_t)128 * N * 2;   // 1,572,864
    float*    mA = (float*)   (ws + off); off += (size_t)N * 128 * 4;   // 3,145,728
    float*    mB = (float*)   (ws + off); off += (size_t)N * 128 * 4;   // 3,145,728
    _Float16* zs = (_Float16*)(ws + off); off += (size_t)N * 64 * 2;    //   786,432
    float*    z0 = (float*)   (ws + off); off += (size_t)N * 4;         //    24,576

    float* z_out = out + (size_t)N * N;   // z (N,65) lives after adj_pred

    // layer 0
    expmap_kernel<<<N, 128, 0, stream>>>(x, h0);
    lin_norm_kernel<<<N, 128, 0, stream>>>(h0, 129, 128, 128, W0, b0, s0, 0, gT);
    agg_kernel<<<N / 16, 8 * 32, 0, stream>>>(adj, gT, mA, 128);
    // layer 1
    lin_norm_kernel<<<N, 128, 0, stream>>>(mA, 128, 128, 128, W1, b1, s1, 1, gT);
    agg_kernel<<<N / 16, 8 * 32, 0, stream>>>(adj, gT, mB, 128);
    // layer 2 (width 65, padded to 80 = 5 column tiles)
    lin_norm_kernel<<<N, 128, 0, stream>>>(mB, 128, 65, 80, W2, b2, s2, 1, gT);
    agg_kernel<<<N / 16, 5 * 32, 0, stream>>>(adj, gT, mA, 65);
    // decode
    poincare_kernel<<<N, 64, 0, stream>>>(mA, scale, z_out, zs, z0);
    gram_kernel<<<dim3(N / 64, N / 32), 256, 0, stream>>>(zs, z0, out);
}
